// PredictionAttention_42236708389258
// MI455X (gfx1250) — matrix operand order repaired and checked by reading, not compiled, not run on hardware
//
#include <hip/hip_runtime.h>
#include <hip/hip_bf16.h>
#include <math.h>

// ---------------- problem constants ----------------
#define B_   512
#define C_   32
#define T_   9
#define L_   36
#define H_   8
#define HD_  4
#define MROWS (B_*T_*L_)     // 165888 rows, divisible by 16
#define MTILES (MROWS/16)    // 10368

typedef _Float16 half_t;
typedef __attribute__((ext_vector_type(16))) _Float16 v16h;
typedef __attribute__((ext_vector_type(8)))  _Float16 v8h;
typedef __attribute__((ext_vector_type(8)))  float    v8f;

// ---------------- WMMA tile loaders (gfx1250 wave32 layouts) ----------------
// A 16x32 f16 (MxK): lane<16 -> row=lane, K chunks [0..7],[16..23];
//                    lane>=16 -> row=lane-16, K chunks [8..15],[24..31].
__device__ __forceinline__ v16h wmma_load_a(const half_t* __restrict__ A, int lda,
                                            int row, int kbase, int lane) {
  int hsel = lane >> 4;
  const half_t* p = A + (size_t)row * lda + kbase;
  v8h lo = *(const v8h*)(p + hsel * 8);
  v8h hi = *(const v8h*)(p + 16 + hsel * 8);
  v16h a;
#pragma unroll
  for (int i = 0; i < 8; ++i) { a[i] = lo[i]; a[i + 8] = hi[i]; }
  return a;
}

// B 32x16 f16 (KxN), weights stored [N][K] row-major (native W[out][in]):
// lane n=lane&15 holds K chunks [hsel*8..+7],[16+hsel*8..+7] of column n.
__device__ __forceinline__ v16h wmma_load_b(const half_t* __restrict__ W, int ldw,
                                            int col, int kbase, int lane) {
  int hsel = lane >> 4;
  const half_t* p = W + (size_t)col * ldw + kbase;
  v8h lo = *(const v8h*)(p + hsel * 8);
  v8h hi = *(const v8h*)(p + 16 + hsel * 8);
  v16h b;
#pragma unroll
  for (int i = 0; i < 8; ++i) { b[i] = lo[i]; b[i + 8] = hi[i]; }
  return b;
}

__device__ __forceinline__ v8f wmma32(v16h a, v16h b, v8f c) {
  return __builtin_amdgcn_wmma_f32_16x16x32_f16(false, a, false, b, (short)0, c,
                                                false, false);
}

__device__ __forceinline__ float wave_sum32(float v) {
  v += __shfl_xor(v, 16, 32);
  v += __shfl_xor(v, 8, 32);
  v += __shfl_xor(v, 4, 32);
  v += __shfl_xor(v, 2, 32);
  v += __shfl_xor(v, 1, 32);
  return v;
}

__device__ __forceinline__ int win_start(const int* flagp) {
  int f = flagp[0];
  return (f >= 1 && f <= (L_ / 6 - 1)) ? (L_ - 6 * f) : 0;
}

// ---------------- 1) prep: window splice, NCHW->rows, fp32->f16 ----------------
// xcat[row][0:32]=con, [32:64]=tar ; pin[row][0:32]=con2 (conv input low half)
__global__ void prep_kernel(const float* __restrict__ x, const float* __restrict__ pos,
                            const float* __restrict__ pre, const int* __restrict__ flagp,
                            half_t* __restrict__ xcat, half_t* __restrict__ pin) {
  int idx = blockIdx.x * blockDim.x + threadIdx.x;   // over MROWS*32
  int c = idx & 31;
  int row = idx >> 5;                                 // (b*T+t)*L + l
  int l = row % L_;
  int bt = row / L_;
  int t = bt % T_;
  int b = bt / T_;
  int s = win_start(flagp);
  bool win = (l >= s) && (l < s + 6);
  float xv = x[(((size_t)b * C_ + c) * T_ + t) * L_ + l];
  float pv = pos[((size_t)c * T_ + t) * L_ + l];
  float ev = pre[((size_t)c * T_ + t) * L_ + l];
  float con = win ? pv : xv;
  float tar = win ? xv : pv;
  float con2 = win ? ev : con;
  xcat[(size_t)row * 64 + c]      = (half_t)con;
  xcat[(size_t)row * 64 + 32 + c] = (half_t)tar;
  pin[(size_t)row * 64 + c]       = (half_t)con2;
}

// ---------------- 2) weight repack to f16 (conv OIHW -> O,tap,I) ----------------
__global__ void repack_kernel(const float* __restrict__ kW, const float* __restrict__ qW,
                              const float* __restrict__ kkW, const float* __restrict__ mW,
                              const float* __restrict__ cW,
                              half_t* __restrict__ kWh, half_t* __restrict__ qWh,
                              half_t* __restrict__ kkWh, half_t* __restrict__ mWh,
                              half_t* __restrict__ cWh) {
  int i = blockIdx.x * blockDim.x + threadIdx.x;
  if (i < 2048) kWh[i] = (half_t)kW[i];
  if (i < 1024) {
    qWh[i] = (half_t)qW[i];
    kkWh[i] = (half_t)kkW[i];
    mWh[i] = (half_t)mW[i];
  }
  if (i < 32 * 576) {  // dest: o*576 + tap*64 + ci ; src OIHW: o*576 + ci*9 + tap
    int o = i / 576, r2 = i % 576, tap = r2 / 64, ci = r2 % 64;
    cWh[i] = (half_t)cW[(size_t)o * 576 + ci * 9 + tap];
  }
}

// ---------------- 3) v = l2norm(proj(rule)) ; broadcast over b (324 rows) -------
__global__ void v_kernel(const float* __restrict__ rule, const float* __restrict__ vW,
                         const float* __restrict__ vb, float* __restrict__ vout) {
  int tl = blockIdx.x;           // t*L + l
  int t = tl / L_, l = tl % L_;
  int lane = threadIdx.x;        // output channel
  float acc = vb[lane];
#pragma unroll
  for (int cc = 0; cc < 32; ++cc)
    acc += rule[((size_t)cc * T_ + t) * L_ + l] * vW[(size_t)lane * 32 + cc];
  float ss = acc * acc;
  ss += __shfl_xor(ss, 1, 32);
  ss += __shfl_xor(ss, 2, 32);
  vout[(size_t)tl * 32 + lane] = acc / fmaxf(sqrtf(ss), 1e-12f);
}

// ---------------- 4) generic WMMA GEMM: M=165888, N=32 -------------------------
// out = A(f16, lda) @ W^T(f16,[32][K]) + bias ; f32 or f16 output, ldo, col_off
__global__ void gemm16_kernel(const half_t* __restrict__ A, int lda,
                              const half_t* __restrict__ W, int K,
                              const float* __restrict__ bias,
                              float* __restrict__ outF, half_t* __restrict__ outH,
                              int ldo, int col_off) {
  int lane = threadIdx.x & 31;
  int mtile = blockIdx.x * 4 + (threadIdx.x >> 5);
  int arow = mtile * 16 + (lane & 15);
  v8f acc0 = {0.f, 0.f, 0.f, 0.f, 0.f, 0.f, 0.f, 0.f};
  v8f acc1 = acc0;
  for (int kb = 0; kb < K; kb += 32) {
    v16h a  = wmma_load_a(A, lda, arow, kb, lane);
    v16h b0 = wmma_load_b(W, K, (lane & 15), kb, lane);
    v16h b1 = wmma_load_b(W, K, 16 + (lane & 15), kb, lane);
    acc0 = wmma32(a, b0, acc0);
    acc1 = wmma32(a, b1, acc1);
  }
  int hsel = lane >> 4, n = lane & 15;
#pragma unroll
  for (int r = 0; r < 8; ++r) {
    int orow = mtile * 16 + r + 8 * hsel;
    float v0 = acc0[r] + bias[n];
    float v1 = acc1[r] + bias[16 + n];
    if (outF) {
      outF[(size_t)orow * ldo + col_off + n] = v0;
      outF[(size_t)orow * ldo + col_off + 16 + n] = v1;
    } else {
      outH[(size_t)orow * ldo + col_off + n] = (half_t)v0;
      outH[(size_t)orow * ldo + col_off + 16 + n] = (half_t)v1;
    }
  }
}

// ---------------- 5) tc = l2norm_head(gelu(LN(h))) ------------------------------
__global__ void tc_kernel(const float* __restrict__ h, const float* __restrict__ g,
                          const float* __restrict__ bb, half_t* __restrict__ tc) {
  int row = blockIdx.x * 8 + (threadIdx.x >> 5);
  int lane = threadIdx.x & 31;
  float v = h[(size_t)row * 32 + lane];
  float mu = wave_sum32(v) * (1.f / 32.f);
  float d = v - mu;
  float var = wave_sum32(d * d) * (1.f / 32.f);
  float hn = d * rsqrtf(var + 1e-5f) * g[lane] + bb[lane];
  float ge = 0.5f * hn * (1.f + erff(hn * 0.70710678118654752f));
  float ss = ge * ge;
  ss += __shfl_xor(ss, 1, 32);
  ss += __shfl_xor(ss, 2, 32);
  tc[(size_t)row * 32 + lane] = (half_t)(ge / fmaxf(sqrtf(ss), 1e-12f));
}

// ---------------- 6) q/k finish: per-head l2norm then *= tc ---------------------
__global__ void qkfin_kernel(float* __restrict__ qk, const half_t* __restrict__ tc) {
  int row = blockIdx.x * 8 + (threadIdx.x >> 5);
  int lane = threadIdx.x & 31;
  float v = qk[(size_t)row * 32 + lane];
  float ss = v * v;
  ss += __shfl_xor(ss, 1, 32);
  ss += __shfl_xor(ss, 2, 32);
  qk[(size_t)row * 32 + lane] =
      (v / fmaxf(sqrtf(ss), 1e-12f)) * (float)tc[(size_t)row * 32 + lane];
}

// ---------------- 7) fused attention per (b,t): HD=4 on VALU --------------------
__global__ void attn_kernel(const float* __restrict__ qb, const float* __restrict__ kb,
                            const float* __restrict__ vtl, half_t* __restrict__ rbuf) {
  int bt = blockIdx.x;
  int t = bt % T_;
  __shared__ float qs[L_ * 32], ks2[L_ * 32], vs[L_ * 32];
  for (int i = threadIdx.x; i < L_ * 32; i += blockDim.x) {
    qs[i] = qb[(size_t)bt * L_ * 32 + i];
    ks2[i] = kb[(size_t)bt * L_ * 32 + i];
    vs[i] = vtl[(size_t)t * L_ * 32 + i];
  }
  __syncthreads();
  int tid = threadIdx.x;
  if (tid < H_ * L_) {
    int hh = tid / L_, l = tid % L_;
    const float* qv = qs + l * 32 + hh * 4;
    float q0 = qv[0], q1 = qv[1], q2 = qv[2], q3 = qv[3];
    float mx = -1e30f;
    for (int m = 0; m < L_; ++m) {
      const float* kv = ks2 + m * 32 + hh * 4;
      float sc = (q0 * kv[0] + q1 * kv[1] + q2 * kv[2] + q3 * kv[3]) * 0.5f;
      mx = fmaxf(mx, sc);
    }
    float den = 0.f, r0 = 0.f, r1 = 0.f, r2 = 0.f, r3 = 0.f;
    for (int m = 0; m < L_; ++m) {
      const float* kv = ks2 + m * 32 + hh * 4;
      float sc = (q0 * kv[0] + q1 * kv[1] + q2 * kv[2] + q3 * kv[3]) * 0.5f;
      float e = expf(sc - mx);
      den += e;
      const float* vv = vs + m * 32 + hh * 4;
      r0 += e * vv[0]; r1 += e * vv[1]; r2 += e * vv[2]; r3 += e * vv[3];
    }
    float inv = 1.f / den;
    half_t* o = rbuf + (size_t)bt * L_ * 32 + (size_t)l * 32 + hh * 4;
    o[0] = (half_t)(r0 * inv); o[1] = (half_t)(r1 * inv);
    o[2] = (half_t)(r2 * inv); o[3] = (half_t)(r3 * inv);
  }
}

// ---------------- 8) implicit-GEMM 3x3 conv 64->32 + BN + ReLU (WMMA) -----------
// K = 9 taps * 64ch = 576 = 18 k-steps of 32. A gathered from pin (channels-last).
__global__ void conv_kernel(const half_t* __restrict__ pin, const half_t* __restrict__ Wc,
                            const float* __restrict__ cb, const float* __restrict__ bng,
                            const float* __restrict__ bnb, float* __restrict__ y) {
  int lane = threadIdx.x & 31;
  int mtile = blockIdx.x * 4 + (threadIdx.x >> 5);
  int row = mtile * 16 + (lane & 15);
  int l = row % L_;
  int bt = row / L_;
  int t = bt % T_;
  int b = bt / T_;
  int hsel = lane >> 4;
  v8f acc0 = {0.f, 0.f, 0.f, 0.f, 0.f, 0.f, 0.f, 0.f};
  v8f acc1 = acc0;
  for (int ks = 0; ks < 18; ++ks) {
    int tap = ks >> 1;
    int chb = (ks & 1) * 32;
    int tt = t + tap / 3 - 1;
    int ll = l + tap % 3 - 1;
    v16h a;
#pragma unroll
    for (int i = 0; i < 16; ++i) a[i] = (_Float16)0.f;
    if (tt >= 0 && tt < T_ && ll >= 0 && ll < L_) {
      const half_t* p = pin + (((size_t)b * T_ + tt) * L_ + ll) * 64 + chb;
      v8h lo = *(const v8h*)(p + hsel * 8);
      v8h hi = *(const v8h*)(p + 16 + hsel * 8);
#pragma unroll
      for (int i = 0; i < 8; ++i) { a[i] = lo[i]; a[i + 8] = hi[i]; }
    }
    v16h b0 = wmma_load_b(Wc, 576, (lane & 15), ks * 32, lane);
    v16h b1 = wmma_load_b(Wc, 576, 16 + (lane & 15), ks * 32, lane);
    acc0 = wmma32(a, b0, acc0);
    acc1 = wmma32(a, b1, acc1);
  }
  int n = lane & 15;
#pragma unroll
  for (int r = 0; r < 8; ++r) {
    int orow = mtile * 16 + r + 8 * hsel;
    float v0 = fmaxf((acc0[r] + cb[n]) * bng[n] + bnb[n], 0.f);
    float v1 = fmaxf((acc1[r] + cb[16 + n]) * bng[16 + n] + bnb[16 + n], 0.f);
    y[(size_t)orow * 32 + n] = v0;
    y[(size_t)orow * 32 + 16 + n] = v1;
  }
}

// ---------------- 9) p-linear (L->6) + window reassembly ------------------------
__global__ void final_kernel(const float* __restrict__ x, const float* __restrict__ y,
                             const float* __restrict__ pW, const float* __restrict__ pb,
                             const int* __restrict__ flagp, float* __restrict__ out) {
  int idx = blockIdx.x * blockDim.x + threadIdx.x;  // over B*C*T*L
  int l = idx % L_;
  int t = (idx / L_) % T_;
  int c = (idx / (L_ * T_)) % C_;
  int b = idx / (L_ * T_ * C_);
  int s = win_start(flagp);
  float val;
  if (l < s || l >= s + 6) {
    val = x[idx];
  } else {
    int o = l - s;
    float acc = pb[o];
#pragma unroll
    for (int ll = 0; ll < L_; ++ll)
      acc += y[((((size_t)b * T_) + t) * L_ + ll) * 32 + c] * pW[o * L_ + ll];
    val = acc;
  }
  out[idx] = val;
}

// ---------------- host launcher ------------------------------------------------
extern "C" void kernel_launch(void* const* d_in, const int* in_sizes, int n_in,
                              void* d_out, int out_size, void* d_ws, size_t ws_size,
                              hipStream_t stream) {
  (void)in_sizes; (void)n_in; (void)out_size; (void)ws_size;
  const float* x        = (const float*)d_in[0];
  const float* pos      = (const float*)d_in[1];
  const float* rule     = (const float*)d_in[2];
  const float* pre      = (const float*)d_in[3];
  const float* kernel_W = (const float*)d_in[4];
  const float* kernel_b = (const float*)d_in[5];
  const float* ln_g     = (const float*)d_in[6];
  const float* ln_b     = (const float*)d_in[7];
  const float* q_W      = (const float*)d_in[8];
  const float* q_b      = (const float*)d_in[9];
  const float* k_W      = (const float*)d_in[10];
  const float* k_b      = (const float*)d_in[11];
  const float* v_W      = (const float*)d_in[12];
  const float* v_b      = (const float*)d_in[13];
  const float* m_W      = (const float*)d_in[14];
  const float* m_b      = (const float*)d_in[15];
  const float* conv_W   = (const float*)d_in[16];
  const float* conv_b   = (const float*)d_in[17];
  const float* bn_g     = (const float*)d_in[18];
  const float* bn_b     = (const float*)d_in[19];
  const float* p_W      = (const float*)d_in[20];
  const float* p_b      = (const float*)d_in[21];
  const int*   flag     = (const int*)d_in[22];
  float* out = (float*)d_out;

  char* ws = (char*)d_ws;
  size_t off = 0;
  auto carve = [&](size_t bytes) -> char* {
    char* p = ws + off;
    off = (off + bytes + 255) & ~(size_t)255;
    return p;
  };
  half_t* xcat = (half_t*)carve((size_t)MROWS * 64 * 2);   // [con|tar] f16
  float*  hbuf = (float*) carve((size_t)MROWS * 32 * 4);   // h; later reused as conv y
  half_t* tc   = (half_t*)carve((size_t)MROWS * 32 * 2);
  float*  qbuf = (float*) carve((size_t)MROWS * 32 * 4);
  float*  kbuf = (float*) carve((size_t)MROWS * 32 * 4);
  half_t* rbuf = (half_t*)carve((size_t)MROWS * 32 * 2);
  half_t* pin  = (half_t*)carve((size_t)MROWS * 64 * 2);   // [con2|r@m] f16
  float*  vbuf = (float*) carve((size_t)T_ * L_ * 32 * 4);
  half_t* kWh  = (half_t*)carve(32 * 64 * 2);
  half_t* qWh  = (half_t*)carve(32 * 32 * 2);
  half_t* kkWh = (half_t*)carve(32 * 32 * 2);
  half_t* mWh  = (half_t*)carve(32 * 32 * 2);
  half_t* cWh  = (half_t*)carve(32 * 576 * 2);

  // 1) prep + 2) repack + 3) v
  prep_kernel<<<MROWS * 32 / 256, 256, 0, stream>>>(x, pos, pre, flag, xcat, pin);
  repack_kernel<<<(32 * 576 + 255) / 256, 256, 0, stream>>>(kernel_W, q_W, k_W, m_W,
                                                            conv_W, kWh, qWh, kkWh,
                                                            mWh, cWh);
  v_kernel<<<T_ * L_, 32, 0, stream>>>(rule, v_W, v_b, vbuf);

  // 4) kernel linear (K=64) -> h
  gemm16_kernel<<<MTILES / 4, 128, 0, stream>>>(xcat, 64, kWh, 64, kernel_b,
                                                hbuf, nullptr, 32, 0);
  // 5) tc
  tc_kernel<<<MROWS / 8, 256, 0, stream>>>(hbuf, ln_g, ln_b, tc);
  // q from tar half, k from con half
  gemm16_kernel<<<MTILES / 4, 128, 0, stream>>>(xcat + 32, 64, qWh, 32, q_b,
                                                qbuf, nullptr, 32, 0);
  gemm16_kernel<<<MTILES / 4, 128, 0, stream>>>(xcat, 64, kkWh, 32, k_b,
                                                kbuf, nullptr, 32, 0);
  // 6) l2norm * tc
  qkfin_kernel<<<MROWS / 8, 256, 0, stream>>>(qbuf, tc);
  qkfin_kernel<<<MROWS / 8, 256, 0, stream>>>(kbuf, tc);
  // 7) attention -> r (f16)
  attn_kernel<<<B_ * T_, 320, 0, stream>>>(qbuf, kbuf, vbuf, rbuf);
  // 8) m projection straight into conv input channels 32..63 (f16)
  gemm16_kernel<<<MTILES / 4, 128, 0, stream>>>(rbuf, 32, mWh, 32, m_b,
                                                nullptr, pin, 64, 32);
  // 9) conv(+BN+ReLU) -> y (reuses hbuf)
  conv_kernel<<<MTILES / 4, 128, 0, stream>>>(pin, cWh, conv_b, bn_g, bn_b, hbuf);
  // 10) p-linear + splice
  final_kernel<<<MROWS * 32 / 256, 256, 0, stream>>>(x, hbuf, p_W, p_b, flag, out);
}
